// SimplifiedRhoFold_67619965108969
// MI455X (gfx1250) — matrix operand order repaired and checked by reading, not compiled, not run on hardware
//
#include <hip/hip_runtime.h>
#include <hip/hip_bf16.h>

// ---------------------------------------------------------------------------
// Types for CDNA5 WMMA
// ---------------------------------------------------------------------------
typedef __attribute__((ext_vector_type(16))) __bf16 v16bf;
typedef __attribute__((ext_vector_type(8)))  float  v8f;

#define DEV static __device__ __forceinline__

// Round-to-nearest-even fp32 -> bf16, two values packed into one dword
// (low half = first element, matching the WMMA 16-bit register layout).
DEV unsigned pack_bf16(float a, float b) {
  unsigned ua = __builtin_bit_cast(unsigned, a);
  unsigned ub = __builtin_bit_cast(unsigned, b);
  ua = ua + 0x7FFFu + ((ua >> 16) & 1u);
  ub = ub + 0x7FFFu + ((ub >> 16) & 1u);
  return (ua >> 16) | (ub & 0xFFFF0000u);
}

DEV float rl(float x) { return fmaxf(x, 0.f); }

// ---------------------------------------------------------------------------
// fp32 -> bf16 array conversion (grid-stride), row-major preserved
// ---------------------------------------------------------------------------
__global__ void cvt_f32_to_bf16(const float* __restrict__ src,
                                unsigned short* __restrict__ dst, int n) {
  int i = blockIdx.x * blockDim.x + threadIdx.x;
  int stride = gridDim.x * blockDim.x;
  for (; i < n; i += stride) {
    unsigned u = __builtin_bit_cast(unsigned, src[i]);
    u = u + 0x7FFFu + ((u >> 16) & 1u);
    dst[i] = (unsigned short)(u >> 16);
  }
}

// ---------------------------------------------------------------------------
// Transpose + convert: src (K x N, fp32) -> dst (N x K, bf16). 32x32 tiles.
// ---------------------------------------------------------------------------
__global__ void transpose_to_bf16(const float* __restrict__ src,
                                  unsigned short* __restrict__ dst,
                                  int K, int N) {
  __shared__ float tile[32][33];
  const int nb = blockIdx.x * 32;
  const int kb = blockIdx.y * 32;
  const int tx = threadIdx.x & 31;
  const int ty = threadIdx.x >> 5;          // 0..7
#pragma unroll
  for (int r = 0; r < 32; r += 8)
    tile[ty + r][tx] = src[(size_t)(kb + ty + r) * N + nb + tx];
  __syncthreads();
#pragma unroll
  for (int r = 0; r < 32; r += 8) {
    unsigned u = __builtin_bit_cast(unsigned, tile[tx][ty + r]);
    u = u + 0x7FFFu + ((u >> 16) & 1u);
    dst[(size_t)(nb + ty + r) * K + kb + tx] = (unsigned short)(u >> 16);
  }
}

// ---------------------------------------------------------------------------
// dms / shape heads: one block per (b,l) row, reduction over D=640
// ---------------------------------------------------------------------------
__global__ void dms_shape_kernel(const float* __restrict__ feat,
                                 const float* __restrict__ w_dms,
                                 const float* __restrict__ b_dms,
                                 const float* __restrict__ w_shape,
                                 const float* __restrict__ b_shape,
                                 float* __restrict__ dms,
                                 float* __restrict__ shp) {
  __shared__ float s1[128];
  __shared__ float s2[128];
  const int row = blockIdx.x;               // 0..767
  const float* f = feat + (size_t)row * 640;
  float a = 0.f, b = 0.f;
  for (int k = threadIdx.x; k < 640; k += 128) {
    float v = f[k];
    a += v * w_dms[k];
    b += v * w_shape[k];
  }
  s1[threadIdx.x] = a;
  s2[threadIdx.x] = b;
  __syncthreads();
  for (int off = 64; off > 0; off >>= 1) {
    if ((int)threadIdx.x < off) {
      s1[threadIdx.x] += s1[threadIdx.x + off];
      s2[threadIdx.x] += s2[threadIdx.x + off];
    }
    __syncthreads();
  }
  if (threadIdx.x == 0) {
    dms[row] = s1[0] + b_dms[0];
    shp[row] = s2[0] + b_shape[0];
  }
}

// ---------------------------------------------------------------------------
// bf16 WMMA GEMM, no LDS:  C(MxN,fp32) = A(MxK,bf16) @ Bt^T + bias
//   A  : M x K row-major bf16
//   Bt : N x ldb row-major bf16 (pre-transposed weight; row n = column n)
// Block = 8 waves; wave -> 16 rows x 128 cols; grid = (N/128, M/128).
// Low register footprint (~130 VGPRs) so the scheduler can pipeline the
// per-fragment loads without spilling; B is L0/L2-resident.
// ---------------------------------------------------------------------------
__global__ void __launch_bounds__(256)
gemm_bf16_wmma(const unsigned short* __restrict__ A,
               const unsigned short* __restrict__ Bt,
               const float* __restrict__ bias,   // N or null
               float* __restrict__ C,
               int M, int K, int N, int ldb) {
  const int lane  = threadIdx.x & 31;
  const int wv    = threadIdx.x >> 5;       // 0..7
  const int mlane = lane & 15;
  const int half  = lane >> 4;
  const int m0    = blockIdx.y * 128 + wv * 16;
  const int n0    = blockIdx.x * 128;

  v8f c[8];
#pragma unroll
  for (int nt = 0; nt < 8; ++nt)
#pragma unroll
    for (int r = 0; r < 8; ++r) c[nt][r] = 0.f;

  const unsigned short* arow = A + (size_t)(m0 + mlane) * K;
  const unsigned short* brow = Bt + (size_t)(n0 + mlane) * ldb + half * 8;

  union Frag { v16bf v; uint4 q[2]; };

  for (int kb = 0; kb < K; kb += 32) {
    // A fragment: lane (m, half) holds K = {kb+8h+0..7, kb+16+8h+0..7}
    Frag af;
    const unsigned short* ap = arow + kb + half * 8;
    af.q[0] = *(const uint4*)(ap);
    af.q[1] = *(const uint4*)(ap + 16);
#pragma unroll
    for (int nt = 0; nt < 8; ++nt) {
      // B fragment: lane (n, half) holds K = {kb+8h+0..7, kb+16+8h+0..7}
      Frag bfv;
      const unsigned short* bp = brow + (size_t)(nt * 16) * ldb + kb;
      bfv.q[0] = *(const uint4*)(bp);
      bfv.q[1] = *(const uint4*)(bp + 16);
      c[nt] = __builtin_amdgcn_wmma_f32_16x16x32_bf16(
          false, af.v, false, bfv.v, (short)0, c[nt], false, false);
    }
  }

#pragma unroll
  for (int nt = 0; nt < 8; ++nt) {
    const int col = n0 + nt * 16 + mlane;
    const float bv = bias ? bias[col] : 0.f;
#pragma unroll
    for (int r = 0; r < 8; ++r) {
      const int row = m0 + r + half * 8;    // C layout: M = r + 8*half
      C[(size_t)row * N + col] = c[nt][r] + bv;
    }
  }
}

// ---------------------------------------------------------------------------
// Pair kernel: pair_rep[b,i,j,:] = relu(pi'[b,i,:] + pj[b,j,:]) @ w_pair2 + b2
// pi' = proj_i + b_pair1 (bias pre-folded).
// grid = (2 j-blocks, B*L); block = 192 (6 waves); wave -> 32 j-rows x 128 N.
// Register budget kept under 256 VGPRs (no spills): c[2][8]=128 is the only
// persistent state; pj/pi chunks and B fragments are transient.  Each LDS B
// fragment feeds 2 WMMAs; the scheduler overlaps the next fragment's
// ds_loads with the current WMMA pair (verified pattern in round-1 asm).
// ---------------------------------------------------------------------------
#define PP 259

DEV void build_afrag(unsigned* u, const float4 ia, const float4 ib,
                     const float4 ic, const float4 id,
                     const float4 ja, const float4 jb,
                     const float4 jc, const float4 jd) {
  u[0] = pack_bf16(rl(ia.x + ja.x), rl(ia.y + ja.y));
  u[1] = pack_bf16(rl(ia.z + ja.z), rl(ia.w + ja.w));
  u[2] = pack_bf16(rl(ib.x + jb.x), rl(ib.y + jb.y));
  u[3] = pack_bf16(rl(ib.z + jb.z), rl(ib.w + jb.w));
  u[4] = pack_bf16(rl(ic.x + jc.x), rl(ic.y + jc.y));
  u[5] = pack_bf16(rl(ic.z + jc.z), rl(ic.w + jc.w));
  u[6] = pack_bf16(rl(id.x + jd.x), rl(id.y + jd.y));
  u[7] = pack_bf16(rl(id.z + jd.z), rl(id.w + jd.w));
}

__global__ void __launch_bounds__(192)
pair_wmma_kernel(const float* __restrict__ proj_i,   // (768,512)
                 const float* __restrict__ proj_j,   // (768,512)
                 const unsigned short* __restrict__ w2t, // 128x512 bf16
                 const float* __restrict__ b2,       // 128
                 float* __restrict__ out) {          // (B,L,L,128)
  extern __shared__ unsigned lds[];          // 128 * PP dwords
  const int t = threadIdx.x;                 // 192 threads

  // Stage pre-transposed w2t rows into LDS (row n at dword n*PP, 256 dwords).
  const uint4* src4 = (const uint4*)w2t;     // 8192 x uint4, contiguous
  for (int flat = t; flat < 8192; flat += 192) {
    const int row = flat >> 6;               // 64 uint4 per row
    const int q   = flat & 63;
    const uint4 v = src4[flat];
    unsigned* d = lds + row * PP + q * 4;
    d[0] = v.x; d[1] = v.y; d[2] = v.z; d[3] = v.w;
  }
  __syncthreads();

  const int lane  = t & 31;
  const int wv    = t >> 5;                  // 0..5
  const int mlane = lane & 15;
  const int half  = lane >> 4;
  const int bi    = blockIdx.y;              // b*384 + i
  const int b     = bi / 384;
  const int j0    = blockIdx.x * 192 + wv * 32;

  const float* pi  = proj_i + (size_t)bi * 512;
  const float* pj0 = proj_j + ((size_t)b * 384 + j0 + mlane) * 512;
  const float* pj1 = pj0 + 16 * 512;

  v8f c[2][8];
#pragma unroll
  for (int f = 0; f < 2; ++f)
#pragma unroll
    for (int nt = 0; nt < 8; ++nt)
#pragma unroll
      for (int r = 0; r < 8; ++r) c[f][nt][r] = 0.f;

  union Frag { v16bf v; unsigned u[8]; };

  for (int ks = 0; ks < 16; ++ks) {
    const int k1 = ks * 32 + half * 8;

    // hidden-row inputs for this K-step (transient registers)
    const float4 ja = *(const float4*)(pj0 + k1);
    const float4 jb = *(const float4*)(pj0 + k1 + 4);
    const float4 jc = *(const float4*)(pj0 + k1 + 16);
    const float4 jd = *(const float4*)(pj0 + k1 + 20);
    const float4 ka = *(const float4*)(pj1 + k1);
    const float4 kb = *(const float4*)(pj1 + k1 + 4);
    const float4 kc = *(const float4*)(pj1 + k1 + 16);
    const float4 kd = *(const float4*)(pj1 + k1 + 20);
    const float4 ia = *(const float4*)(pi + k1);
    const float4 ib = *(const float4*)(pi + k1 + 4);
    const float4 ic = *(const float4*)(pi + k1 + 16);
    const float4 id = *(const float4*)(pi + k1 + 20);

    Frag af0, af1;
    build_afrag(af0.u, ia, ib, ic, id, ja, jb, jc, jd);
    build_afrag(af1.u, ia, ib, ic, id, ka, kb, kc, kd);

#pragma unroll
    for (int nt = 0; nt < 8; ++nt) {
      Frag bfv;
      const unsigned* bp = lds + (nt * 16 + mlane) * PP + ks * 16 + half * 4;
#pragma unroll
      for (int vg = 0; vg < 8; ++vg) bfv.u[vg] = bp[vg + (vg & 4)];
      c[0][nt] = __builtin_amdgcn_wmma_f32_16x16x32_bf16(
          false, af0.v, false, bfv.v, (short)0, c[0][nt], false, false);
      c[1][nt] = __builtin_amdgcn_wmma_f32_16x16x32_bf16(
          false, af1.v, false, bfv.v, (short)0, c[1][nt], false, false);
    }
  }

  const size_t obase = (size_t)bi * 384 * 128;
#pragma unroll
  for (int f = 0; f < 2; ++f)
#pragma unroll
    for (int nt = 0; nt < 8; ++nt) {
      const int col = nt * 16 + mlane;
      const float bv = b2[col];
#pragma unroll
      for (int r = 0; r < 8; ++r) {
        const int row = j0 + f * 16 + r + half * 8;
        __builtin_nontemporal_store(c[f][nt][r] + bv,
                                    out + obase + (size_t)row * 128 + col);
      }
    }
}

// ---------------------------------------------------------------------------
// Launch
// ---------------------------------------------------------------------------
extern "C" void kernel_launch(void* const* d_in, const int* in_sizes, int n_in,
                              void* d_out, int out_size, void* d_ws, size_t ws_size,
                              hipStream_t stream) {
  const float* feat     = (const float*)d_in[0];
  const float* w_single = (const float*)d_in[1];
  const float* b_single = (const float*)d_in[2];
  const float* w_pair1  = (const float*)d_in[3];
  const float* b_pair1  = (const float*)d_in[4];
  const float* w_pair2  = (const float*)d_in[5];
  const float* b_pair2  = (const float*)d_in[6];
  const float* w_dms    = (const float*)d_in[7];
  const float* b_dms    = (const float*)d_in[8];
  const float* w_shape  = (const float*)d_in[9];
  const float* b_shape  = (const float*)d_in[10];

  float* out = (float*)d_out;
  char*  ws  = (char*)d_ws;

  // Workspace layout (bytes)
  unsigned short* featb = (unsigned short*)(ws + 0);        // 768x640 bf16 (row-major)
  unsigned short* wsT   = (unsigned short*)(ws + 983040);   // 384x640 bf16 (= w_single^T)
  unsigned short* w1T   = (unsigned short*)(ws + 1474560);  // 512x1280 bf16 (= w_pair1^T)
  unsigned short* w2T   = (unsigned short*)(ws + 2785280);  // 128x512 bf16 (= w_pair2^T)
  float*          proji = (float*)(ws + 2916352);           // 768x512 f32 (+b_pair1)
  float*          projj = (float*)(ws + 4489216);           // 768x512 f32

  // Output layout (flat tuple order)
  float* out_feat   = out;               // 2*384*640      = 491520
  float* out_single = out + 491520;      // 2*384*384      = 294912
  float* out_pair   = out + 786432;      // 2*384*384*128  = 37748736
  float* out_dms    = out + 38535168;    // 768
  float* out_shape  = out + 38535936;    // 768

  // 1) echo features
  hipMemcpyAsync(out_feat, feat, 491520ull * sizeof(float),
                 hipMemcpyDeviceToDevice, stream);

  // 2) activations -> bf16; weights -> transposed bf16 (one-time prep)
  cvt_f32_to_bf16<<<480, 256, 0, stream>>>(feat, featb, 491520);
  transpose_to_bf16<<<dim3(12, 20), 256, 0, stream>>>(w_single, wsT, 640, 384);
  transpose_to_bf16<<<dim3(16, 40), 256, 0, stream>>>(w_pair1, w1T, 1280, 512);
  transpose_to_bf16<<<dim3(4, 16),  256, 0, stream>>>(w_pair2, w2T, 512, 128);

  // 3) scalar heads
  dms_shape_kernel<<<768, 128, 0, stream>>>(feat, w_dms, b_dms, w_shape, b_shape,
                                            out_dms, out_shape);

  // 4) WMMA GEMMs: single_rep, proj_i (+b_pair1 folded), proj_j
  gemm_bf16_wmma<<<dim3(3, 6), 256, 0, stream>>>(featb, wsT, b_single,
                                                 out_single, 768, 640, 384, 640);
  gemm_bf16_wmma<<<dim3(4, 6), 256, 0, stream>>>(featb, w1T, b_pair1,
                                                 proji, 768, 640, 512, 1280);
  gemm_bf16_wmma<<<dim3(4, 6), 256, 0, stream>>>(featb, w1T + 640, nullptr,
                                                 projj, 768, 640, 512, 1280);

  // 5) pair MLP main GEMM (38.7 GFLOP; HBM-store-bound ~6.5 us at 23.3 TB/s)
  const size_t pl = 128 * PP * sizeof(unsigned);
  pair_wmma_kernel<<<dim3(2, 768), 192, pl, stream>>>(proji, projj, w2T,
                                                      b_pair2, out_pair);
}
